// LNKillingRelu_94489280610
// MI455X (gfx1250) — matrix operand order repaired and checked by reading, not compiled, not run on hardware
//
#include <hip/hip_runtime.h>

// LNKillingRelu fused pipeline for MI455X (gfx1250, wave32, WMMA).
//
//  K1a: W(256x256 f32)  -> W_hi/W_lo bf16 (row-major, f contiguous)
//  K1b: x(b,f,c,n f32)  -> x_hi/x_lo bf16 TRANSPOSED to [b][c][n][f]   (f contiguous)
//  K2 : fused  D = W@x  (v_wmma_f32_16x16x32_bf16, 3-term hi/lo split = ~fp32 acc),
//       kf = 6 * x^T G d  over c (G hardcoded sparse Killing form),
//       out = kf<0 ? x : x + kf*d   -- all in-register per wave.
//       Epilogue fp32 x slab is staged via async global->LDS DMA (ASYNCcnt)
//       issued at kernel entry so the whole WMMA loop hides its HBM latency.
//
// Workspace layout (d_ws): x_hi 128MiB | x_lo 128MiB | W_hi 128KiB | W_lo 128KiB
// (requires ws_size >= ~269 MB)

typedef __attribute__((ext_vector_type(16))) __bf16 v16bf;
typedef __attribute__((ext_vector_type(8)))  __bf16 v8bf;
typedef __attribute__((ext_vector_type(8)))  float  v8f;
typedef __attribute__((ext_vector_type(4)))  float  v4f;

#define B_DIM 8
#define F_DIM 256
#define C_DIM 8
#define N_DIM 4096

#define XHI_OFF 0ull
#define XLO_OFF 134217728ull                 // 8*256*8*4096 * 2B
#define WHI_OFF 268435456ull
#define WLO_OFF (268435456ull + 131072ull)

static __device__ inline v16bf cat16(v8bf a, v8bf b) {
  return __builtin_shufflevector(a, b, 0,1,2,3,4,5,6,7,8,9,10,11,12,13,14,15);
}

// Async global->LDS B128 copy (tracked by ASYNCcnt; no VGPR data movement).
// lds_off = wave-relative LDS byte offset (low 32 bits of generic pointer,
// ISA section 10.2: flat->LDS truncates to addr[31:0]).
static __device__ inline void async_ld_b128(unsigned lds_off, const void* g) {
  asm volatile("global_load_async_to_lds_b128 %0, %1, off"
               :: "v"(lds_off), "v"(g) : "memory");
}

static __device__ inline void wait_async0() {
#if __has_builtin(__builtin_amdgcn_s_wait_asynccnt)
  __builtin_amdgcn_s_wait_asynccnt(0);
#else
  asm volatile("s_wait_asynccnt 0x0" ::: "memory");
#endif
  asm volatile("" ::: "memory");  // compiler ordering for subsequent LDS reads
}

// ---------------------------------------------------------------- K1a: W split
__global__ __launch_bounds__(256)
void k_split_w(const float* __restrict__ W, __bf16* __restrict__ whi,
               __bf16* __restrict__ wlo) {
  const size_t base = ((size_t)blockIdx.x * 256 + threadIdx.x) * 8;
  v4f a = *(const v4f*)(W + base);
  v4f b = *(const v4f*)(W + base + 4);
  float vals[8];
#pragma unroll
  for (int e = 0; e < 4; ++e) { vals[e] = a[e]; vals[4 + e] = b[e]; }
  v8bf hi, lo;
#pragma unroll
  for (int e = 0; e < 8; ++e) {
    __bf16 h = (__bf16)vals[e];
    hi[e] = h;
    lo[e] = (__bf16)(vals[e] - (float)h);
  }
  *(v8bf*)(whi + base) = hi;
  *(v8bf*)(wlo + base) = lo;
}

// ------------------------------------------------- K1b: x split + transpose
// Per block: one 64(f) x 64(n) tile of one (b,c) plane, via LDS transpose.
// Output layout: [b][c][n][f] bf16, f contiguous (pitch 256).
__global__ __launch_bounds__(256)
void k_split_x(const float* __restrict__ x, __bf16* __restrict__ xhiT,
               __bf16* __restrict__ xloT) {
  __shared__ float tile[64][65];
  const int t  = threadIdx.x;
  const int bc = blockIdx.z;               // b*8 + c
  const int n0 = blockIdx.x * 64;
  const int f0 = blockIdx.y * 64;
  const int nl = t & 63;
  const int fb = (t >> 6) * 16;
#pragma unroll
  for (int i = 0; i < 16; ++i) {
    const int fl = fb + i;
    tile[fl][nl] =
        x[(((size_t)(bc >> 3) * F_DIM + f0 + fl) * C_DIM + (bc & 7)) * N_DIM +
          n0 + nl];
  }
  __syncthreads();
  const int fseg = (t & 7) * 8;
#pragma unroll
  for (int j = 0; j < 2; ++j) {
    const int nl2 = (t >> 3) + j * 32;
    v8bf hi, lo;
#pragma unroll
    for (int e = 0; e < 8; ++e) {
      float v   = tile[fseg + e][nl2];
      __bf16 h  = (__bf16)v;
      hi[e]     = h;
      lo[e]     = (__bf16)(v - (float)h);
    }
    const size_t o = ((size_t)bc * N_DIM + n0 + nl2) * F_DIM + f0 + fseg;
    *(v8bf*)(xhiT + o) = hi;
    *(v8bf*)(xloT + o) = lo;
  }
}

// ------------------------------------------------------------------ K2: fused
// Block = 512 threads = 16 waves; wave w owns g-rows [w*16, w*16+16).
// Grid = (N/16, B). Each wave: 8 fp32 accumulators (one per c) of one 16x16
// (g,n) tile; K=256 swept in 8 steps of 32 with 3 bf16 WMMAs per (k,c).
__global__ __launch_bounds__(512)
void k_fused(const __bf16* __restrict__ xhiT, const __bf16* __restrict__ xloT,
             const __bf16* __restrict__ whi,  const __bf16* __restrict__ wlo,
             const float* __restrict__ x, float* __restrict__ out) {
  // Per-wave epilogue slab: [wave][g_local][c][n] fp32, 8KB/wave = 128KB/block.
  __shared__ __align__(16) float xbuf[16][16][8][16];

  const int n0   = blockIdx.x * 16;
  const int b    = blockIdx.y;
  const int wave = threadIdx.x >> 5;
  const int lane = threadIdx.x & 31;
  const int lr   = lane & 15;   // row (A) / col (B) index inside tile
  const int half = lane >> 4;   // K-half selector
  const int g0   = wave * 16;

  // Fire async DMA of this wave's epilogue fp32 x slab (16g x 8c x 16n = 8KB)
  // into LDS: 512 x 16B chunks = 16 wave-instructions. Consumed after the
  // WMMA loop behind a single s_wait_asynccnt 0 (wave-local; no barrier).
#pragma unroll
  for (int i = 0; i < 16; ++i) {
    const int chunk = i * 32 + lane;     // 0..511
    const int row   = chunk >> 2;        // g_local*8 + c
    const int q     = chunk & 3;         // 16B quarter of the 64B row
    const int gl    = row >> 3;
    const int c     = row & 7;
    const float* gsrc =
        x + ((size_t)(b * F_DIM + g0 + gl) * C_DIM + c) * N_DIM + n0 + q * 4;
    const unsigned ldst = (unsigned)(uintptr_t)&xbuf[wave][gl][c][q * 4];
    async_ld_b128(ldst, gsrc);
  }

  v8f acc[8] = {};  // acc[c] : 16x16 fp32 D tile in WMMA C/D layout

  // A fragment (16x32 bf16): lane<16 holds K {k0..k0+7} and {k0+16..k0+23};
  // lane>=16 holds {k0+8..k0+15} and {k0+24..k0+31}.  (row pitch 256 elem)
  const size_t arow = (size_t)(g0 + lr) * F_DIM;
  // B fragment (32x16 bf16): same K split as A (lane<16 {k0..k0+7},{k0+16..k0+23}; lane>=16 {k0+8..k0+15},{k0+24..k0+31}).
  const size_t brow = ((size_t)(b * C_DIM) * N_DIM + (size_t)(n0 + lr)) * F_DIM;

  for (int k = 0; k < 8; ++k) {
    const int    k0 = k * 32;
    const size_t ao = arow + k0 + half * 8;
    v16bf Ah = cat16(*(const v8bf*)(whi + ao), *(const v8bf*)(whi + ao + 16));
    v16bf Al = cat16(*(const v8bf*)(wlo + ao), *(const v8bf*)(wlo + ao + 16));
#pragma unroll
    for (int c = 0; c < 8; ++c) {
      const size_t bo = brow + (size_t)c * (N_DIM * F_DIM) + k0 + half * 8;
      v16bf Bh = cat16(*(const v8bf*)(xhiT + bo), *(const v8bf*)(xhiT + bo + 16));
      v16bf Bl = cat16(*(const v8bf*)(xloT + bo), *(const v8bf*)(xloT + bo + 16));
      // D += Wh*Xh + Wh*Xl + Wl*Xh  (fp32 accumulate; l*l ~ 2^-18 term dropped)
      acc[c] = __builtin_amdgcn_wmma_f32_16x16x32_bf16(
          false, Ah, false, Bh, (short)0, acc[c], false, false);
      acc[c] = __builtin_amdgcn_wmma_f32_16x16x32_bf16(
          false, Ah, false, Bl, (short)0, acc[c], false, false);
      acc[c] = __builtin_amdgcn_wmma_f32_16x16x32_bf16(
          false, Al, false, Bh, (short)0, acc[c], false, false);
    }
  }

  // Wait for the async staging DMA, then run the epilogue out of LDS.
  wait_async0();

  // Epilogue: C/D layout => VGPR j, lane => (g = g0 + j + 8*half, n = n0 + lr).
  // All 8 c-values of d live in the same lane/slot across acc[0..7].
  const int nidx = n0 + lr;
#pragma unroll
  for (int j = 0; j < 8; ++j) {
    const int    gl   = j + 8 * half;
    const size_t base = ((size_t)(b * F_DIM + g0 + gl) * C_DIM) * N_DIM + nidx;
    float xv[8], dv[8];
#pragma unroll
    for (int c = 0; c < 8; ++c) {
      xv[c] = xbuf[wave][gl][c][lr];
      dv[c] = acc[c][j];
    }
    // Killing-form Gram matrix of the sl(3) basis (sparse, hardcoded):
    // G00=G11=2, G01=G10=1, G24=G42=G36=G63=G57=G75=1.
    float kf = 2.f * xv[0] * dv[0] + xv[0] * dv[1] + xv[1] * dv[0] +
               2.f * xv[1] * dv[1] + xv[2] * dv[4] + xv[4] * dv[2] +
               xv[3] * dv[6] + xv[6] * dv[3] + xv[5] * dv[7] + xv[7] * dv[5];
    kf *= 6.f;
#pragma unroll
    for (int c = 0; c < 8; ++c) {
      const float o = (kf < 0.f) ? xv[c] : fmaf(kf, dv[c], xv[c]);
      out[base + (size_t)c * N_DIM] = o;
    }
  }
}

// ------------------------------------------------------------------- launcher
extern "C" void kernel_launch(void* const* d_in, const int* in_sizes, int n_in,
                              void* d_out, int out_size, void* d_ws,
                              size_t ws_size, hipStream_t stream) {
  (void)in_sizes; (void)n_in; (void)out_size; (void)ws_size;
  const float* x = (const float*)d_in[0];
  const float* W = (const float*)d_in[1];
  float* out     = (float*)d_out;
  char*  ws      = (char*)d_ws;

  __bf16* xhi = (__bf16*)(ws + XHI_OFF);
  __bf16* xlo = (__bf16*)(ws + XLO_OFF);
  __bf16* whi = (__bf16*)(ws + WHI_OFF);
  __bf16* wlo = (__bf16*)(ws + WLO_OFF);

  // K1a: 256*256 / (256 thr * 8 elem) = 32 blocks
  k_split_w<<<32, 256, 0, stream>>>(W, whi, wlo);
  // K1b: (n tiles, f tiles, b*c) = (4096/64, 256/64, 64)
  k_split_x<<<dim3(64, 4, 64), 256, 0, stream>>>(x, xhi, xlo);
  // K2: (N/16, B) blocks of 512 threads (16 waves cover all 256 g rows)
  k_fused<<<dim3(N_DIM / 16, B_DIM), 512, 0, stream>>>(xhi, xlo, whi, wlo, x,
                                                       out);
}
